// DMS_STAttention_48722109006190
// MI455X (gfx1250) — matrix operand order repaired and checked by reading, not compiled, not run on hardware
//
#include <hip/hip_runtime.h>

typedef __attribute__((ext_vector_type(16))) _Float16 v16h;
typedef __attribute__((ext_vector_type(8)))  _Float16 v8h;
typedef __attribute__((ext_vector_type(8)))  float    v8f;

#define C_DIM 128
#define T_DIM 10
#define J_DIM 22
#define ROWS  220          // T*J rows per batch slab
#define ROWS_PAD 224       // padded to 14 tiles of 16
#define AS_STRIDE 136      // halfs per row (128 + 8 pad -> bank-conflict break)
#define W_STRIDE  144      // halfs per W row (128 + 16 pad)

#define AS_BYTES   (ROWS_PAD * AS_STRIDE * 2)        // 60,928
#define WL_BYTES   (2 * 128 * W_STRIDE * 2)          // 73,728
#define DACC_OFF   (AS_BYTES + WL_BYTES)             // 134,656 (16B aligned)
#define WSM_OFF    (DACC_OFF + 448 * 4)              // 136,448
#define AVL_OFF    (WSM_OFF + 448 * 4)               // 138,240
#define LDS_BYTES  (AVL_OFF + 256 * 4)               // 139,264 (2 WGs / WGP)

__global__ __launch_bounds__(256)
void dms_stattn_kernel(const float* __restrict__ src,
                       const float* __restrict__ Ws,
                       const float* __restrict__ ads,   // a_dst_s [128]
                       const float* __restrict__ Wt,
                       const float* __restrict__ adt,   // a_dst_t [128]
                       const float* __restrict__ sb,    // sa_bias [22*22]
                       const float* __restrict__ tb,    // ta_bias [10*10]
                       float* __restrict__ out,
                       int nB)
{
    extern __shared__ char smem[];
    _Float16* As   = (_Float16*)smem;                   // [ROWS_PAD][AS_STRIDE], row = t*22+j
    _Float16* Wl   = (_Float16*)(smem + AS_BYTES);      // [2][128][W_STRIDE] f16 of W[h][c]
    float*    dacc = (float*)(smem + DACC_OFF);         // [2][224] d-logit results
    float*    wsm  = (float*)(smem + WSM_OFF);          // [2][224] softmax weights
    float*    avl  = (float*)(smem + AVL_OFF);          // [2][128] a_dst vectors

    const int tid = threadIdx.x;
    const int b   = blockIdx.x;
    const float* srcb = src + (size_t)b * (C_DIM * ROWS);

    // ---------------- Phase 0: stage slab (transposed) + weights into LDS as f16 ---------
    // src is streamed exactly once per block: non-temporal loads keep L2 for W/biases.
    for (int idx = tid; idx < C_DIM * ROWS; idx += 256) {
        int c  = idx / ROWS;
        int tj = idx - c * ROWS;
        As[tj * AS_STRIDE + c] = (_Float16)__builtin_nontemporal_load(srcb + idx);
    }
    for (int idx = tid; idx < 4 * C_DIM; idx += 256) {      // zero pad rows 220..223
        int r = ROWS + (idx >> 7);
        As[r * AS_STRIDE + (idx & 127)] = (_Float16)0.0f;
    }
    for (int idx = tid; idx < 128 * C_DIM; idx += 256) {
        int h = idx >> 7, c = idx & 127;
        Wl[h * W_STRIDE + c]                  = (_Float16)Ws[idx];
        Wl[128 * W_STRIDE + h * W_STRIDE + c] = (_Float16)Wt[idx];
    }
    if (tid < 128)       avl[tid]       = ads[tid];
    else if (tid < 256)  avl[tid]       = adt[tid - 128];
    __syncthreads();

    // ---------------- Phase 1: per-wave strip = (branch, m_tile); sweep all 8 n_tiles ----
    // d_row = sum_h lrelu(x[row,h]) * a_dst[h] accumulated in registers across n_tiles;
    // one cross-lane butterfly + plain LDS store per strip (exclusive row ownership).
    const int wave = tid >> 5;
    const int lane = tid & 31;
    const int ln   = lane & 15;
    const int hiH  = lane >> 4;                 // 0: lanes 0-15, 1: lanes 16-31
    const int klo  = hiH ? 16 : 0;              // A-fragment K sub-offset
    const int kbo  = hiH ? 16 : 0;              // B-fragment K sub-offset

    for (int unit = wave; unit < 28; unit += 8) {
        const int branch = (unit >= 14);        // 0 = spatial (Ws), 1 = temporal (Wt)
        const int m_tile = branch ? unit - 14 : unit;

        // A row for this lane (M = ln); temporal branch permutes rows j*10+t -> t*22+j
        int lane_m = m_tile * 16 + ln;
        int src_row = lane_m;
        if (branch && lane_m < ROWS) {
            int j = lane_m / T_DIM;
            int t = lane_m - j * T_DIM;
            src_row = t * J_DIM + j;
        }
        const _Float16* arow = As + src_row * AS_STRIDE;

        // Hoist the strip's A fragments (all 4 K-steps) into registers
        union frag { v16h v; v8h h[2]; };
        frag af[4];
        #pragma unroll
        for (int kk = 0; kk < 4; ++kk) {
            af[kk].h[0] = *(const v8h*)(arow + kk * 32 + klo);
            af[kk].h[1] = *(const v8h*)(arow + kk * 32 + klo + 8);
        }

        const _Float16* wbr  = Wl + branch * (128 * W_STRIDE);
        const float*    avec = avl + branch * 128;

        float pr[8];
        #pragma unroll
        for (int r = 0; r < 8; ++r) pr[r] = 0.0f;

        for (int n_tile = 0; n_tile < 8; ++n_tile) {
            const _Float16* wbase = wbr + (n_tile * 16 + ln) * W_STRIDE;
            v8f acc = {};
            #pragma unroll
            for (int kk = 0; kk < 4; ++kk) {
                frag bf;
                bf.h[0] = *(const v8h*)(wbase + kk * 32 + kbo);
                bf.h[1] = *(const v8h*)(wbase + kk * 32 + kbo + 8);
                acc = __builtin_amdgcn_wmma_f32_16x16x32_f16(
                          false, af[kk].v, false, bf.v, (short)0, acc, false, false);
            }
            // fuse LeakyReLU + weight by a_dst[h] (h = this lane's column)
            const float aval = avec[n_tile * 16 + ln];
            #pragma unroll
            for (int r = 0; r < 8; ++r) {
                float v = acc[r];
                v = (v >= 0.0f) ? v : 0.2f * v;
                pr[r] += v * aval;
            }
        }

        // one butterfly over the 16 columns (lanes sharing hiH half) per strip
        #pragma unroll
        for (int m = 1; m <= 8; m <<= 1) {
            #pragma unroll
            for (int r = 0; r < 8; ++r) pr[r] += __shfl_xor(pr[r], m, 32);
        }
        if (ln == 0) {
            // D layout: VGPR r -> row m_tile*16 + hiH*8 + r ; strip owns its rows
            float* dd = dacc + branch * 224 + m_tile * 16 + hiH * 8;
            #pragma unroll
            for (int r = 0; r < 8; ++r) dd[r] = pr[r];
        }
    }
    __syncthreads();

    // ---------------- Phase 2a: softmax over each group's d (s_i cancels out) -----------
    if (tid < T_DIM) {                          // spatial: group t, softmax over j (22)
        int t = tid;
        float vals[J_DIM], mx = -1e30f;
        for (int j = 0; j < J_DIM; ++j) { vals[j] = dacc[t * J_DIM + j]; mx = fmaxf(mx, vals[j]); }
        float s = 0.0f;
        for (int j = 0; j < J_DIM; ++j) { vals[j] = __expf(vals[j] - mx); s += vals[j]; }
        float inv = 1.0f / s;
        for (int j = 0; j < J_DIM; ++j) wsm[t * J_DIM + j] = vals[j] * inv;
    } else if (tid >= 32 && tid < 32 + J_DIM) { // temporal: group j, softmax over t (10)
        int j = tid - 32;
        float vals[T_DIM], mx = -1e30f;
        for (int t = 0; t < T_DIM; ++t) { vals[t] = dacc[224 + j * T_DIM + t]; mx = fmaxf(mx, vals[t]); }
        float s = 0.0f;
        for (int t = 0; t < T_DIM; ++t) { vals[t] = __expf(vals[t] - mx); s += vals[t]; }
        float inv = 1.0f / s;
        for (int t = 0; t < T_DIM; ++t) wsm[224 + j * T_DIM + t] = vals[t] * inv;
    }
    __syncthreads();

    // ---------------- Phase 2b: broadcast rows + bias; streaming (NT) stores -------------
    const size_t SA_TOTAL = (size_t)nB * (T_DIM * J_DIM * J_DIM);
    float* osa = out + (size_t)b * (T_DIM * J_DIM * J_DIM);   // [T][J][J] contiguous
    for (int idx = tid; idx < T_DIM * J_DIM * J_DIM; idx += 256) {
        int t = idx / (J_DIM * J_DIM);
        int rem = idx - t * (J_DIM * J_DIM);                  // rem = i*22 + j
        int j = rem % J_DIM;
        __builtin_nontemporal_store(wsm[t * J_DIM + j] + sb[rem], osa + idx);
    }
    float* ota = out + SA_TOTAL + (size_t)b * (J_DIM * T_DIM * T_DIM); // [J][T][T]
    for (int idx = tid; idx < J_DIM * T_DIM * T_DIM; idx += 256) {
        int j = idx / (T_DIM * T_DIM);
        int rem = idx - j * (T_DIM * T_DIM);                  // rem = i*10 + t'
        int t2 = rem % T_DIM;
        __builtin_nontemporal_store(wsm[224 + j * T_DIM + t2] + tb[rem], ota + idx);
    }
    if (b == 0 && tid < 4) {
        out[SA_TOTAL + (size_t)nB * (J_DIM * T_DIM * T_DIM) + tid] = 0.0f;
    }
}

extern "C" void kernel_launch(void* const* d_in, const int* in_sizes, int n_in,
                              void* d_out, int out_size, void* d_ws, size_t ws_size,
                              hipStream_t stream) {
    const float* src = (const float*)d_in[0];
    const float* Ws  = (const float*)d_in[1];
    // d_in[2] = a_src_s: cancels in softmax along axis=-1, unused
    const float* ads = (const float*)d_in[3];
    const float* Wt  = (const float*)d_in[4];
    // d_in[5] = a_src_t: unused (cancels)
    const float* adt = (const float*)d_in[6];
    const float* sb  = (const float*)d_in[7];
    const float* tb  = (const float*)d_in[8];
    float* out = (float*)d_out;

    int nB = in_sizes[0] / (C_DIM * T_DIM * J_DIM);   // 4096

    dms_stattn_kernel<<<nB, 256, LDS_BYTES, stream>>>(
        src, Ws, ads, Wt, adt, sb, tb, out, nB);
}
